// MTRB_10350871183515
// MI455X (gfx1250) — compile-verified, matrix operand order checked by reading, not run on hardware
//
#include <hip/hip_runtime.h>
#include <hip/hip_bf16.h>
#include <math.h>

// ---------- types ----------
typedef __attribute__((ext_vector_type(16))) __bf16          bf16x16;
typedef __attribute__((ext_vector_type(16))) unsigned short  u16x16;
typedef __attribute__((ext_vector_type(8)))  unsigned short  u16x8;
typedef __attribute__((ext_vector_type(8)))  float           f32x8;

// ---------- helpers ----------
__device__ inline unsigned short f2bf(float f) {
    union { float f; unsigned u; } v; v.f = f;
    unsigned r = v.u + 0x7FFFu + ((v.u >> 16) & 1u);   // round-to-nearest-even
    return (unsigned short)(r >> 16);
}
__device__ inline float bf2f(unsigned short b) {
    union { unsigned u; float f; } v; v.u = ((unsigned)b) << 16; return v.f;
}
__device__ inline float loadf(const float* p, size_t i) { return p[i]; }
__device__ inline float loadf(const unsigned short* p, size_t i) { return bf2f(p[i]); }
__device__ inline int clampi(int v, int lo, int hi) { return v < lo ? lo : (v > hi ? hi : v); }

__device__ inline f32x8 wmma_bf16(u16x16 a, u16x16 b, f32x8 c) {
    return __builtin_amdgcn_wmma_f32_16x16x32_bf16(
        false, __builtin_bit_cast(bf16x16, a),
        false, __builtin_bit_cast(bf16x16, b),
        (short)0, c, false, false);
}

// K index held by element i of a v16bf fragment in a given lane (ISA 7.12.2,
// 16-bit A 16x32 layout; B mirrored with N = lane%16).
__device__ inline int frag_k(int i, int lane) {
    return (i < 8 ? i : i + 8) + (lane < 16 ? 0 : 8);
}

// ---------- weight swizzle: OIHW f32 -> per-lane bf16 B fragments ----------
// layout: [kt][nt][lane 0..31][i 0..15], K padded per 32-channel group.
__global__ void swz_weights(const float* __restrict__ w, unsigned short* __restrict__ out,
                            int Cin, int Cout, int khw, int ntiles, int ktotal) {
    int idx = blockIdx.x * blockDim.x + threadIdx.x;
    int total = ktotal * ntiles * 512;
    if (idx >= total) return;
    int i    = idx & 15;
    int lane = (idx >> 4) & 31;
    int nt   = (idx >> 9) % ntiles;
    int kt   = (idx >> 9) / ntiles;

    int kb = frag_k(i, lane);
    int n  = nt * 16 + (lane & 15);
    int nfull = Cin >> 5;
    int gi, ksg;
    if (kt < nfull * khw) { gi = kt / khw; ksg = kt % khw; }
    else                  { gi = nfull;    ksg = kt - nfull * khw; }
    int kig = ksg * 32 + kb;
    int cl  = kig / khw, r = kig % khw;
    int cin = gi * 32 + cl;
    float v = 0.f;
    if (cin < Cin && n < Cout && cl < 32)
        v = w[((size_t)n * Cin + cin) * khw + r];
    out[idx] = f2bf(v);
}

// ---------- implicit-GEMM 3x3 conv (pad 1, stride 1) with WMMA bf16 ----------
// block = 128 threads (4 waves). Each block: 64 pixels of one row x all Cout.
// LDS holds a per-pixel im2col tile [pixel 0..63][k 0..287] (stride 296 halves)
// so every A fragment is two ds_load_b128's of 8 consecutive K values.
// act: 0 none, 1 GELU(exact), 2 ReLU.  resid (f32, out-shaped) optional.
template<typename T, int NT>
__global__ void conv3x3_wmma(const T* __restrict__ in, const unsigned short* __restrict__ bswz,
                             const float* __restrict__ bias, const float* __restrict__ resid,
                             float* __restrict__ out, int Cin, int Cout, int act) {
    const int H = 256, W = 256, LSTR = 296;              // 296*2 B row stride (16B aligned, bank-spread)
    __shared__ __align__(16) unsigned short lds2[64 * LSTR];   // 37.9 KB
    int w0 = blockIdx.x * 64;
    int h  = blockIdx.y;
    int nb = blockIdx.z;
    int tid = threadIdx.x, lane = tid & 31, wv = tid >> 5;

    f32x8 acc[NT];
    #pragma unroll
    for (int t = 0; t < NT; ++t)
        #pragma unroll
        for (int j = 0; j < 8; ++j) acc[t][j] = 0.f;

    int nfull = Cin >> 5, rem = Cin & 31;
    int ngroups = nfull + (rem ? 1 : 0);
    int ktail = rem ? (rem * 9 + 31) / 32 : 0;

    int p_lane = wv * 16 + (lane & 15);                  // this lane's output pixel (M row)
    int kbase  = (lane < 16) ? 0 : 8;

    for (int gi = 0; gi < ngroups; ++gi) {
        int c0 = gi * 32;
        int ks_n = (gi < nfull) ? 9 : ktail;
        int kmax = ks_n * 32;

        __syncthreads();
        // stage im2col tile: chunk = (k-octet, pixel); consecutive lanes -> consecutive w (coalesced)
        int nchunk = 64 * (kmax >> 3);
        for (int ci = tid; ci < nchunk; ci += 128) {
            int kc = ci >> 6;            // k-octet index
            int p  = ci & 63;            // pixel
            int k0 = kc << 3;
            u16x8 vv;
            #pragma unroll
            for (int i = 0; i < 8; ++i) {
                int k = k0 + i;
                int cin = c0 + k / 9;
                int r = k % 9;
                int kh = r / 3, kw = r % 3;
                int hh = h + kh - 1;
                int ww = w0 + p + kw - 1;
                float v = 0.f;
                if (cin < Cin && (unsigned)hh < (unsigned)H && (unsigned)ww < (unsigned)W)
                    v = loadf(in, (((size_t)nb * Cin + cin) * H + hh) * W + ww);
                vv[i] = f2bf(v);
            }
            *(u16x8*)&lds2[p * LSTR + k0] = vv;          // ds_store_b128
        }
        // prefetch next channel group's input rows while we compute on this one
        if (gi + 1 < ngroups) {
            int cpf = c0 + 32 + (tid & 31);
            if (cpf < Cin)
                __builtin_prefetch(&in[(((size_t)nb * Cin + cpf) * H + h) * W + w0 + (tid >> 5) * 16], 0, 0);
        }
        __syncthreads();

        for (int ks = 0; ks < ks_n; ++ks) {
            int ktg = gi * 9 + ks;
            int off = p_lane * LSTR + ks * 32 + kbase;
            u16x8 lo = *(const u16x8*)&lds2[off];        // K = ks*32 + kbase + 0..7
            u16x8 hi = *(const u16x8*)&lds2[off + 16];   // K = ks*32 + 16 + kbase + 0..7
            u16x16 a;
            #pragma unroll
            for (int i = 0; i < 8; ++i) { a[i] = lo[i]; a[i + 8] = hi[i]; }
            #pragma unroll
            for (int nt = 0; nt < NT; ++nt) {
                const u16x16* bp = (const u16x16*)(bswz + ((size_t)(ktg * NT + nt) * 32 + lane) * 16);
                acc[nt] = wmma_bf16(a, *bp, acc[nt]);
            }
        }
    }

    // epilogue: D layout -> lane holds col n=lane%16, rows m = v + (lane<16?0:8)
    int m_row = (lane < 16) ? 0 : 8;
    int n_sub = lane & 15;
    #pragma unroll
    for (int nt = 0; nt < NT; ++nt) {
        int cout = nt * 16 + n_sub;
        if (cout >= Cout) continue;
        float bb = bias ? bias[cout] : 0.f;
        #pragma unroll
        for (int v = 0; v < 8; ++v) {
            int mm = v + m_row;
            int wq = w0 + wv * 16 + mm;
            float x = acc[nt][v] + bb;
            if (act == 1)      x = 0.5f * x * (1.f + erff(x * 0.70710678f));
            else if (act == 2) x = fmaxf(x, 0.f);
            size_t oi = (((size_t)nb * Cout + cout) * H + h) * W + wq;
            if (resid) x += resid[oi];
            out[oi] = x;
        }
    }
}

// ---------- depthwise 3x3, stride 8, pad 2, dilation 2 ----------
__global__ void depthwise_s8(const float* __restrict__ y1, const float* __restrict__ dwc_w,
                             const float* __restrict__ dwc_b, float* __restrict__ y2) {
    int idx = blockIdx.x * blockDim.x + threadIdx.x;
    if (idx >= 2 * 64 * 32 * 32) return;
    int ow = idx & 31, oh = (idx >> 5) & 31, c = (idx >> 10) & 63, nb = idx >> 16;
    float acc = dwc_b[c];
    #pragma unroll
    for (int kh = 0; kh < 3; ++kh) {
        int ih = oh * 8 - 2 + 2 * kh;
        if ((unsigned)ih >= 256u) continue;
        #pragma unroll
        for (int kw = 0; kw < 3; ++kw) {
            int iw = ow * 8 - 2 + 2 * kw;
            if ((unsigned)iw >= 256u) continue;
            acc += y1[(((size_t)nb * 64 + c) * 256 + ih) * 256 + iw] * dwc_w[c * 9 + kh * 3 + kw];
        }
    }
    y2[idx] = acc;
}

// ---------- per-8x8-block DCT then IDCT (explicit, as in reference) ----------
__global__ void dct_idct(const float* __restrict__ y2, float* __restrict__ y3) {
    int b = blockIdx.x * blockDim.x + threadIdx.x;
    if (b >= 2 * 64 * 4 * 4) return;
    int bw = b & 3, bh = (b >> 2) & 3, c = (b >> 4) & 63, nb = b >> 10;
    const float* src = y2 + (((size_t)nb * 64 + c) * 32 + bh * 8) * 32 + bw * 8;
    float*       dst = y3 + (((size_t)nb * 64 + c) * 32 + bh * 8) * 32 + bw * 8;

    const float PI = 3.14159265358979f;
    float D[8][8];
    for (int i = 0; i < 8; ++i)
        for (int j = 0; j < 8; ++j)
            D[i][j] = (i == 0) ? 0.3535533906f : 0.5f * cosf(PI * (2 * j + 1) * i / 16.f);

    float X[8][8];
    for (int i = 0; i < 8; ++i)
        for (int j = 0; j < 8; ++j) X[i][j] = src[i * 32 + j];

    float T1[8][8], CB[8][8], T2[8][8];
    for (int i = 0; i < 8; ++i)                     // T1 = D * X
        for (int j = 0; j < 8; ++j) { float s = 0.f; for (int k = 0; k < 8; ++k) s += D[i][k] * X[k][j]; T1[i][j] = s; }
    for (int i = 0; i < 8; ++i)                     // CB = T1 * D^T
        for (int j = 0; j < 8; ++j) { float s = 0.f; for (int k = 0; k < 8; ++k) s += T1[i][k] * D[j][k]; CB[i][j] = s; }
    for (int i = 0; i < 8; ++i)                     // T2 = D^T * CB
        for (int j = 0; j < 8; ++j) { float s = 0.f; for (int k = 0; k < 8; ++k) s += D[k][i] * CB[k][j]; T2[i][j] = s; }
    for (int i = 0; i < 8; ++i)                     // out = T2 * D
        for (int j = 0; j < 8; ++j) { float s = 0.f; for (int k = 0; k < 8; ++k) s += T2[i][k] * D[k][j]; dst[i * 32 + j] = s; }
}

// ---------- fused: gate = sigmoid(1x1 conv via WMMA), x8 bilinear upsample, ----------
// ---------- multiply, write bf16 into gated[:, br*64:(br+1)*64]             ----------
__global__ void gate_upsample(const float* __restrict__ t, const unsigned short* __restrict__ dw3swz,
                              const float* __restrict__ dw3_b, const float* __restrict__ y3,
                              unsigned short* __restrict__ gated, int br) {
    const int H = 256, W = 256;
    __shared__ unsigned short lds[32 * 64];
    int w0 = blockIdx.x * 64, h = blockIdx.y, nb = blockIdx.z;
    int tid = threadIdx.x, lane = tid & 31, wv = tid >> 5;

    for (int idx = tid; idx < 2048; idx += 128) {
        int c = idx >> 6, col = idx & 63;
        lds[idx] = f2bf(t[(((size_t)nb * 32 + c) * H + h) * W + w0 + col]);
    }
    __syncthreads();

    int m = lane & 15, col = wv * 16 + m;
    u16x16 a;
    #pragma unroll
    for (int i = 0; i < 16; ++i) {
        int k = frag_k(i, lane);         // K = 32 exactly: one WMMA step
        a[i] = lds[k * 64 + col];
    }

    float sh = h * 0.125f - 0.4375f;
    float h0f = floorf(sh), fh = sh - h0f;
    int h0 = (int)h0f;
    int h0c = clampi(h0, 0, 31), h1c = clampi(h0 + 1, 0, 31);

    int n_sub = lane & 15, m_row = (lane < 16) ? 0 : 8;
    #pragma unroll
    for (int nt = 0; nt < 4; ++nt) {
        f32x8 acc;
        #pragma unroll
        for (int j = 0; j < 8; ++j) acc[j] = 0.f;
        const u16x16* bp = (const u16x16*)(dw3swz + ((size_t)(br * 4 + nt) * 32 + lane) * 16);
        acc = wmma_bf16(a, *bp, acc);

        int cout = nt * 16 + n_sub;                 // within 64-channel slice
        float bb = dw3_b[br * 64 + cout];
        const float* Y = y3 + ((size_t)(nb * 64 + cout)) * 32 * 32;
        #pragma unroll
        for (int v = 0; v < 8; ++v) {
            int mm = v + m_row;
            int wq = w0 + wv * 16 + mm;
            float g = acc[v] + bb;
            g = 1.f / (1.f + expf(-g));
            float sw = wq * 0.125f - 0.4375f;
            float wf = floorf(sw), fw = sw - wf;
            int iw0 = (int)wf;
            int wc0 = clampi(iw0, 0, 31), wc1 = clampi(iw0 + 1, 0, 31);
            float y00 = Y[h0c * 32 + wc0], y01 = Y[h0c * 32 + wc1];
            float y10 = Y[h1c * 32 + wc0], y11 = Y[h1c * 32 + wc1];
            float val = (1.f - fh) * ((1.f - fw) * y00 + fw * y01)
                      +        fh  * ((1.f - fw) * y10 + fw * y11);
            val *= g;
            gated[(((size_t)nb * 512 + br * 64 + cout) * H + h) * W + wq] = f2bf(val);
        }
    }
}

// ---------- host ----------
extern "C" void kernel_launch(void* const* d_in, const int* in_sizes, int n_in,
                              void* d_out, int out_size, void* d_ws, size_t ws_size,
                              hipStream_t stream) {
    const float* x      = (const float*)d_in[0];
    const float* conv_w = (const float*)d_in[1];
    const float* conv_b = (const float*)d_in[2];
    const float* ddct_w = (const float*)d_in[3];
    const float* ddct_b = (const float*)d_in[4];
    const float* dctc_w = (const float*)d_in[5];
    const float* dctc_b = (const float*)d_in[6];
    const float* dw1_w  = (const float*)d_in[7];
    const float* dw1_b  = (const float*)d_in[8];
    const float* dw3_w  = (const float*)d_in[9];
    const float* dw3_b  = (const float*)d_in[10];
    const float* ar_w   = (const float*)d_in[11];
    const float* ar_b   = (const float*)d_in[12];
    float* out = (float*)d_out;

    char* ws = (char*)d_ws;
    size_t off = 0;
    auto alloc = [&](size_t bytes) -> char* {
        char* p = ws + off; off = (off + bytes + 255) & ~(size_t)255; return p;
    };
    float* feat = (float*)alloc(2ull * 64 * 256 * 256 * 4);
    float* tbuf = (float*)alloc(2ull * 32 * 256 * 256 * 4);
    float* y1   = (float*)alloc(2ull * 64 * 256 * 256 * 4);
    float* y2   = (float*)alloc(2ull * 64 * 32 * 32 * 4);
    float* y3   = (float*)alloc(2ull * 64 * 32 * 32 * 4);
    unsigned short* gated     = (unsigned short*)alloc(2ull * 512 * 256 * 256 * 2);
    unsigned short* swz_conv1 = (unsigned short*)alloc(21ull  * 4  * 512 * 2);
    unsigned short* swz_dw1   = (unsigned short*)alloc(21ull  * 2  * 512 * 2);
    unsigned short* swz_ar    = (unsigned short*)alloc(144ull * 5  * 512 * 2);
    unsigned short* swz_dw3   = (unsigned short*)alloc(1ull   * 32 * 512 * 2);
    unsigned short* swz_ddct[7];
    for (int i = 0; i < 7; ++i) swz_ddct[i] = (unsigned short*)alloc(18ull * 4 * 512 * 2);

    auto swzlaunch = [&](const float* w, unsigned short* o, int Cin, int Cout, int khw, int nt, int kt) {
        int total = kt * nt * 512;
        swz_weights<<<(total + 255) / 256, 256, 0, stream>>>(w, o, Cin, Cout, khw, nt, kt);
    };
    swzlaunch(conv_w, swz_conv1,  72,  64, 9,  4,  21);
    swzlaunch(dw1_w,  swz_dw1,    72,  32, 9,  2,  21);
    swzlaunch(ar_w,   swz_ar,    512,  72, 9,  5, 144);
    swzlaunch(dw3_w,  swz_dw3,    32, 512, 1, 32,   1);
    for (int i = 0; i < 7; ++i)
        swzlaunch(ddct_w + (size_t)i * 64 * 64 * 9, swz_ddct[i], 64, 64, 9, 4, 18);

    dim3 cgrid(4, 256, 2);
    dim3 cblk(128);
    // feat = GELU(conv3x3(x))
    conv3x3_wmma<float, 4><<<cgrid, cblk, 0, stream>>>(x, swz_conv1, conv_b, nullptr, feat, 72, 64, 1);
    // t = ReLU(conv3x3(x, dw1))
    conv3x3_wmma<float, 2><<<cgrid, cblk, 0, stream>>>(x, swz_dw1, dw1_b, nullptr, tbuf, 72, 32, 2);

    const int layer_idx[8] = {0, 1, 2, 3, 4, 3, 5, 6};
    for (int br = 0; br < 8; ++br) {
        int li = layer_idx[br];
        conv3x3_wmma<float, 4><<<cgrid, cblk, 0, stream>>>(
            feat, swz_ddct[li], ddct_b + li * 64, feat, y1, 64, 64, 1);
        depthwise_s8<<<(131072 + 255) / 256, 256, 0, stream>>>(
            y1, dctc_w + (size_t)li * 64 * 9, dctc_b + li * 64, y2);
        dct_idct<<<(2048 + 63) / 64, 64, 0, stream>>>(y2, y3);
        gate_upsample<<<cgrid, cblk, 0, stream>>>(tbuf, swz_dw3, dw3_b, y3, gated, br);
    }
    // out = conv3x3(gated bf16, ar_w)
    conv3x3_wmma<unsigned short, 5><<<cgrid, cblk, 0, stream>>>(
        gated, swz_ar, ar_b, nullptr, out, 512, 72, 0);

    (void)in_sizes; (void)n_in; (void)out_size; (void)ws_size;
}
